// MASK_MODEL_23484881174793
// MI455X (gfx1250) — matrix operand order repaired and checked by reading, not compiled, not run on hardware
//
#include <hip/hip_runtime.h>
#include <math.h>
#include <stdint.h>

// ---------------------------------------------------------------------------
// CDNA5 (gfx1250) Mask-RCNN-style pipeline.
//  * Heavy GEMM-shaped stages use v_wmma_f32_16x16x32_f16 (wave32 WMMA) with
//    1 A-fragment x 4 B-fragment register blocking (16x64 tile per wave).
//  * FC GEMM stages its shared 64x32 B tile into LDS with the Tensor Data
//    Mover (tensor_load_to_lds + s_wait_tensorcnt) when the builtin exists;
//    fragments are then read back with ds_load_b128.
// ---------------------------------------------------------------------------

typedef __attribute__((ext_vector_type(16))) _Float16 v16h;
typedef __attribute__((ext_vector_type(8)))  _Float16 v8h;
typedef __attribute__((ext_vector_type(8)))  float    v8f;
typedef __attribute__((ext_vector_type(4)))  unsigned int u32x4;
typedef __attribute__((ext_vector_type(4)))  int i32x4;
typedef __attribute__((ext_vector_type(8)))  int i32x8;

union F16x16 { v16h v; v8h h[2]; };

#if defined(__has_builtin)
#if __has_builtin(__builtin_amdgcn_tensor_load_to_lds)
#define HAVE_TDM 1
#endif
#endif

static constexpr int  kB        = 4;      // batch
static constexpr int  kImg      = 800;
static constexpr int  kFH       = 100;
static constexpr int  kFW       = 100;
static constexpr int  kA        = 3;
static constexpr int  kPreNMS   = 1000;
static constexpr int  kPostNMS  = 100;
static constexpr int  kNumClass = 12;
static constexpr float kNMSThr  = 0.7f;
static constexpr float kBBoxClip = 4.135166556742356f; // ln(1000/16)

#ifdef HAVE_TDM
// 2D TDM load: tile_rows x tile_k f16 elements, row stride = stride_elems,
// from gaddr into LDS byte-offset lds_off (rows packed contiguously).
// D# packing per CDNA5 ISA.
__device__ __forceinline__ void tdm_load_2d_f16(unsigned lds_off,
                                                unsigned long long gaddr,
                                                unsigned tile_k, unsigned tile_rows,
                                                unsigned long long stride_elems,
                                                unsigned tensor_dim0)
{
    u32x4 g0;
    g0[0] = 1u;                                   // count=1, user descriptor
    g0[1] = lds_off;                              // lds_addr (bytes)
    g0[2] = (unsigned)(gaddr & 0xffffffffull);    // global_addr[31:0]
    g0[3] = (unsigned)((gaddr >> 32) & 0x01ffffffull) | (2u << 30); // ga[56:32] | type=2
    i32x8 g1;
    g1[0] = (int)(1u << 16);                      // wg_mask=0, data_size=1 (2 bytes)
    g1[1] = (int)((tensor_dim0 & 0xffffu) << 16); // barrier addr=0 | tensor_dim0[15:0]
    g1[2] = (int)(((tensor_dim0 >> 16) & 0xffffu) |
                  ((tile_rows & 0xffffu) << 16)); // tensor_dim0[31:16] | tensor_dim1[15:0]
    g1[3] = (int)((tile_k & 0xffffu) << 16);      // tensor_dim1[31:16]=0 | tile_dim0
    g1[4] = (int)(tile_rows & 0xffffu);           // tile_dim1 | tile_dim2=0
    g1[5] = (int)(unsigned)(stride_elems & 0xffffffffull);       // dim0_stride[31:0]
    g1[6] = (int)(unsigned)((stride_elems >> 32) & 0xffffull);   // dim0_stride[47:32]
    g1[7] = 0;                                    // dim1_stride unused (2D tile)
    i32x4 gz = {0, 0, 0, 0};
#if __clang_major__ >= 23
    i32x8 gz8 = {0, 0, 0, 0, 0, 0, 0, 0};
    __builtin_amdgcn_tensor_load_to_lds(g0, g1, gz, gz, gz8, 0);
#else
    __builtin_amdgcn_tensor_load_to_lds(g0, g1, gz, gz, 0);
#endif
}
#endif

// ================= f16 WMMA GEMM: C = relu?(A*B^T + bias) ==================
// A: [M][K] f16 row-major; Bm: [N][K] f16 row-major (weights transposed);
// C: [M][N] f16.  K % 32 == 0, N % 64 == 0.
// Block = 256 threads = 8 waves; block tile = 128(M) x 64(N); wave tile 16x64.
// Per K-step the block's shared 64x32 B tile is staged to LDS via TDM.
__global__ void k_gemm_f16(const _Float16* __restrict__ A,
                           const _Float16* __restrict__ Bm,
                           const float* __restrict__ bias,
                           _Float16* __restrict__ C,
                           int M, int N, int K, int relu)
{
    __shared__ _Float16 sB[64 * 32];
    const int tilesN64 = N >> 6;
    const int bm = (int)blockIdx.x / tilesN64;
    const int bn = (int)blockIdx.x % tilesN64;
    const int wid  = threadIdx.x >> 5;
    const int lane = threadIdx.x & 31;
    const int hi = lane >> 4, l16 = lane & 15;

    int rowA = bm * 128 + wid * 16 + l16; if (rowA >= M) rowA = M - 1;
    const _Float16* pa = A + (size_t)rowA * K;
    const _Float16* gB = Bm + (size_t)(bn * 64) * K;

    const v8f vz = {};
    v8f acc[4] = {vz, vz, vz, vz};

    for (int k0 = 0; k0 < K; k0 += 32) {
        __syncthreads();                    // previous tile fully consumed
#ifdef HAVE_TDM
        if (threadIdx.x < 32) {             // wave 0 issues the DMA
            tdm_load_2d_f16((unsigned)(uintptr_t)(const void*)sB,
                            (unsigned long long)(uintptr_t)(gB + k0),
                            32u, 64u, (unsigned long long)K, (unsigned)K);
            __builtin_amdgcn_s_wait_tensorcnt(0);
        }
#else
        for (int i = threadIdx.x; i < 64 * 32; i += 256)
            sB[i] = gB[(size_t)(i >> 5) * K + k0 + (i & 31)];
#endif
        __syncthreads();                    // tile visible to all waves

        if (k0 + 128 < K) __builtin_prefetch(pa + k0 + 128, 0, 1);
        F16x16 af;
        af.h[0] = *(const v8h*)(pa + k0 + 8 * hi);
        af.h[1] = *(const v8h*)(pa + k0 + 16 + 8 * hi);
#pragma unroll
        for (int j = 0; j < 4; ++j) {
            F16x16 bf;
            const _Float16* sr = sB + (size_t)(j * 16 + l16) * 32 + 8 * hi;
            bf.h[0] = *(const v8h*)(sr);
            bf.h[1] = *(const v8h*)(sr + 16);
            acc[j] = __builtin_amdgcn_wmma_f32_16x16x32_f16(false, af.v, false, bf.v,
                                                            (short)0, acc[j], false, false);
        }
    }

#pragma unroll
    for (int j = 0; j < 4; ++j) {
        const int col = bn * 64 + j * 16 + l16;
#pragma unroll
        for (int vv = 0; vv < 8; ++vv) {
            const int mr = bm * 128 + wid * 16 + vv + 8 * hi;
            if (mr < M && col < N) {
                float x = acc[j][vv];
                if (bias) x += bias[col];
                if (relu && x < 0.f) x = 0.f;
                C[(size_t)mr * N + col] = (_Float16)x;
            }
        }
    }
}

// =============== 3x3 conv, implicit GEMM via WMMA (always ReLU) =============
// in : NHWC f16 [Bn][H][W][Cin], Cin % 32 == 0; Cout % 16 == 0.
// w  : f16 [Cout][3][3][Cin]   bias: f32 [Cout] or null
// out: NHWC f16 [Bn][OH][OW][Cout]
// One wave computes a 16(M) x 64(N) tile: A fragment reused by 4 WMMAs.
__global__ void k_conv3x3_wmma(const _Float16* __restrict__ in,
                               const _Float16* __restrict__ w,
                               const float* __restrict__ bias,
                               _Float16* __restrict__ out,
                               int Bn, int H, int W, int Cin,
                               int OH, int OW, int Cout,
                               int stride, int pad)
{
    const int M       = Bn * OH * OW;
    const int ngroups = (Cout + 63) >> 6;
    const int tilesM  = (M + 15) >> 4;
    const int wave    = (int)((blockIdx.x * (size_t)blockDim.x + threadIdx.x) >> 5);
    if (wave >= tilesM * ngroups) return;         // wave-uniform exit
    const int lane = threadIdx.x & 31;
    const int hi   = lane >> 4;
    const int l16  = lane & 15;
    const int tm   = (wave / ngroups) << 4;
    const int ng   = (wave % ngroups) << 6;

    int m = tm + l16; if (m >= M) m = M - 1;
    const int ox_ = m % OW; int t_ = m / OW;
    const int oy_ = t_ % OH; const int b_ = t_ / OH;

    const _Float16* wrow[4];
#pragma unroll
    for (int j = 0; j < 4; ++j) {
        int r = ng + j * 16 + l16; if (r >= Cout) r = Cout - 1;
        wrow[j] = w + (size_t)r * (9 * Cin);
    }

    const v16h vzero = {};
    const v8f vz = {};
    v8f acc[4] = {vz, vz, vz, vz};

    for (int kh = 0; kh < 3; ++kh) {
        const int iy = oy_ * stride + kh - pad;
        for (int kw = 0; kw < 3; ++kw) {
            const int ix = ox_ * stride + kw - pad;
            const bool valid = (iy >= 0) && (iy < H) && (ix >= 0) && (ix < W);
            const int syy = valid ? iy : 0, sxx = valid ? ix : 0;
            const _Float16* arow = in + (((size_t)b_ * H + syy) * W + sxx) * Cin;
            const int kbase = (kh * 3 + kw) * Cin;
            for (int c0 = 0; c0 < Cin; c0 += 32) {
                F16x16 af;
                if (valid) {
                    af.h[0] = *(const v8h*)(arow + c0 + 8 * hi);
                    af.h[1] = *(const v8h*)(arow + c0 + 16 + 8 * hi);
                } else {
                    af.v = vzero;
                }
#pragma unroll
                for (int j = 0; j < 4; ++j) {
                    F16x16 bf;
                    const _Float16* br = wrow[j] + kbase + c0 + 8 * hi;
                    bf.h[0] = *(const v8h*)(br);
                    bf.h[1] = *(const v8h*)(br + 16);
                    acc[j] = __builtin_amdgcn_wmma_f32_16x16x32_f16(false, af.v, false, bf.v,
                                                                    (short)0, acc[j], false, false);
                }
            }
        }
    }

#pragma unroll
    for (int j = 0; j < 4; ++j) {
        const int col = ng + j * 16 + l16;
#pragma unroll
        for (int vv = 0; vv < 8; ++vv) {
            const int mr = tm + vv + 8 * hi;
            if (mr < M && col < Cout) {
                const int ox2 = mr % OW; int tt = mr / OW;
                const int oy2 = tt % OH; const int b2 = tt / OH;
                float x = acc[j][vv];
                if (bias) x += bias[col];
                if (x < 0.f) x = 0.f;
                out[(((size_t)b2 * OH + oy2) * OW + ox2) * Cout + col] = (_Float16)x;
            }
        }
    }
}

// ===================== conv1: 7x7 s2, 3ch, normalize fused ==================
__global__ void k_conv1(const float* __restrict__ img, const float* __restrict__ w,
                        _Float16* __restrict__ out)
{
    const size_t total = (size_t)kB * 400 * 400 * 64;
    size_t tid = blockIdx.x * (size_t)blockDim.x + threadIdx.x;
    if (tid >= total) return;
    const int co = (int)(tid & 63);
    const int ox = (int)((tid >> 6) % 400);
    const int oy = (int)(((tid >> 6) / 400) % 400);
    const int b  = (int)(tid / ((size_t)64 * 400 * 400));
    const float mean[3] = {0.485f, 0.456f, 0.406f};
    const float stdv[3] = {0.229f, 0.224f, 0.225f};
    float s = 0.f;
    for (int ci = 0; ci < 3; ++ci) {
        const float* ip = img + ((size_t)b * 3 + ci) * kImg * kImg;
        const float* wp = w + ((size_t)co * 3 + ci) * 49;
        const float inv = 1.f / stdv[ci];
        for (int kh = 0; kh < 7; ++kh) {
            const int iy = oy * 2 + kh - 2;
            if (iy < 0 || iy >= kImg) continue;
            for (int kw = 0; kw < 7; ++kw) {
                const int ix = ox * 2 + kw - 2;
                if (ix < 0 || ix >= kImg) continue;
                s += (ip[(size_t)iy * kImg + ix] - mean[ci]) * inv * wp[kh * 7 + kw];
            }
        }
    }
    if (s < 0.f) s = 0.f;
    out[tid] = (_Float16)s;   // tid == ((b*400+oy)*400+ox)*64+co  -> NHWC
}

// ============================ weight repacking ==============================
// OIHW f32 [O][I][3][3]  ->  f16 [O][3][3][I]
__global__ void k_pack_conv_w(const float* __restrict__ w, _Float16* __restrict__ o16,
                              int O, int I)
{
    size_t tid = blockIdx.x * (size_t)blockDim.x + threadIdx.x;
    const size_t total = (size_t)O * I * 9;
    if (tid >= total) return;
    const int kk = (int)(tid % 9);
    const int i  = (int)((tid / 9) % I);
    const int o  = (int)(tid / ((size_t)I * 9));
    o16[((size_t)o * 9 + kk) * I + i] = (_Float16)w[((size_t)o * I + i) * 9 + kk];
}

// f32 [K][N] -> f16 [N][K]
__global__ void k_pack_fc_w(const float* __restrict__ w, _Float16* __restrict__ o16,
                            int K, int N)
{
    size_t tid = blockIdx.x * (size_t)blockDim.x + threadIdx.x;
    const size_t total = (size_t)K * N;
    if (tid >= total) return;
    const int n = (int)(tid % N);
    const int k = (int)(tid / N);
    o16[(size_t)n * K + k] = (_Float16)w[(size_t)k * N + n];
}

// deconv w f32 [ci][co][2][2] -> f16 [di*2+dj][ci][co]
__global__ void k_pack_deconv_w(const float* __restrict__ w, _Float16* __restrict__ o16)
{
    size_t tid = blockIdx.x * (size_t)blockDim.x + threadIdx.x;
    if (tid >= (size_t)256 * 256 * 4) return;
    const int dj = (int)(tid & 1);
    const int di = (int)((tid >> 1) & 1);
    const int co = (int)((tid >> 2) & 255);
    const int ci = (int)(tid >> 10);
    o16[(((size_t)(di * 2 + dj) * 256 + ci) * 256) + co] =
        (_Float16)w[(((size_t)ci * 256 + co) * 2 + di) * 2 + dj];
}

// ============================== RPN heads ===================================
__global__ void k_rpn_heads(const _Float16* __restrict__ t,
                            const float* __restrict__ cw, const float* __restrict__ cb,
                            const float* __restrict__ rw, const float* __restrict__ rb,
                            float* __restrict__ scores, float* __restrict__ deltas)
{
    const int total = kB * kA * kFH * kFW;
    int tid = blockIdx.x * blockDim.x + threadIdx.x;
    if (tid >= total) return;
    const int x = tid % kFW;
    const int y = (tid / kFW) % kFH;
    const int a = (tid / (kFH * kFW)) % kA;
    const int b = tid / (kA * kFH * kFW);
    const _Float16* tp = t + (((size_t)b * kFH + y) * kFW + x) * 256;
    float s = cb[a];
    for (int c = 0; c < 256; ++c) s += (float)tp[c] * cw[a * 256 + c];
    const int idx = b * (kA * kFH * kFW) + a * (kFH * kFW) + y * kFW + x;
    scores[idx] = 1.f / (1.f + expf(-s));
    for (int d = 0; d < 4; ++d) {
        float r = rb[a * 4 + d];
        const float* wp = rw + (size_t)(a * 4 + d) * 256;
        for (int c = 0; c < 256; ++c) r += (float)tp[c] * wp[c];
        deltas[(size_t)idx * 4 + d] = r;
    }
}

// ============================ top-k (per batch) =============================
__global__ void k_topk(float* __restrict__ scores, int* __restrict__ top, int n, int k)
{
    const int b = blockIdx.x;
    float* s = scores + (size_t)b * n;
    __shared__ float sv[256];
    __shared__ int   si[256];
    for (int it = 0; it < k; ++it) {
        float best = -1e30f; int bi = n;
        for (int i = threadIdx.x; i < n; i += blockDim.x) {
            const float v = s[i];
            if (v > best || (v == best && i < bi)) { best = v; bi = i; }
        }
        sv[threadIdx.x] = best; si[threadIdx.x] = bi;
        __syncthreads();
        for (int off = 128; off > 0; off >>= 1) {
            if ((int)threadIdx.x < off) {
                if (sv[threadIdx.x + off] > sv[threadIdx.x] ||
                    (sv[threadIdx.x + off] == sv[threadIdx.x] &&
                     si[threadIdx.x + off] < si[threadIdx.x])) {
                    sv[threadIdx.x] = sv[threadIdx.x + off];
                    si[threadIdx.x] = si[threadIdx.x + off];
                }
            }
            __syncthreads();
        }
        if (threadIdx.x == 0) { top[(size_t)b * k + it] = si[0]; s[si[0]] = -1e30f; }
        __syncthreads();
    }
}

// ======================== anchor decode (per proposal) ======================
__global__ void k_decode(const int* __restrict__ top, const float* __restrict__ deltas,
                         float* __restrict__ boxes)
{
    int tid = blockIdx.x * blockDim.x + threadIdx.x;
    if (tid >= kB * kPreNMS) return;
    const int b = tid / kPreNMS;
    const int idx = top[tid];
    const int a = idx / (kFH * kFW);
    const int rem = idx % (kFH * kFW);
    const int y = rem / kFW, x = rem % kFW;
    const float asp[3] = {0.5f, 1.f, 2.f};
    const float sq = sqrtf(asp[a]);
    const float ha = 64.f * sq, wa = 64.f / sq;
    const float cx = (x + 0.5f) * 8.f, cy = (y + 0.5f) * 8.f;
    const float* dp = deltas + ((size_t)b * (kA * kFH * kFW) + idx) * 4;
    const float dx = dp[0], dy = dp[1];
    const float dw = fminf(dp[2], kBBoxClip);
    const float dh = fminf(dp[3], kBBoxClip);
    const float ncx = cx + dx * wa, ncy = cy + dy * ha;
    const float nw = wa * expf(dw), nh = ha * expf(dh);
    float o[4] = { ncx - nw * 0.5f, ncy - nh * 0.5f, ncx + nw * 0.5f, ncy + nh * 0.5f };
    float* bp = boxes + (size_t)tid * 4;
#pragma unroll
    for (int i = 0; i < 4; ++i) bp[i] = fminf(fmaxf(o[i], 0.f), (float)kImg);
}

// =============================== NMS (per batch) ============================
__global__ void k_nms(const float* __restrict__ boxes, float* __restrict__ rois_out)
{
    const int b = blockIdx.x;
    const float* bx = boxes + (size_t)b * kPreNMS * 4;
    __shared__ unsigned char supp[kPreNMS];
    __shared__ unsigned char keep[kPreNMS];
    for (int i = threadIdx.x; i < kPreNMS; i += blockDim.x) { supp[i] = 0; keep[i] = 0; }
    __syncthreads();
    for (int i = 0; i < kPreNMS; ++i) {
        const int ki = supp[i] ? 0 : 1;          // uniform across block
        if (threadIdx.x == 0) keep[i] = (unsigned char)ki;
        if (ki) {
            const float x1 = bx[i * 4 + 0], y1 = bx[i * 4 + 1];
            const float x2 = bx[i * 4 + 2], y2 = bx[i * 4 + 3];
            const float ai = (x2 - x1) * (y2 - y1);
            for (int j = i + 1 + (int)threadIdx.x; j < kPreNMS; j += blockDim.x) {
                const float jx1 = bx[j * 4 + 0], jy1 = bx[j * 4 + 1];
                const float jx2 = bx[j * 4 + 2], jy2 = bx[j * 4 + 3];
                const float aj = (jx2 - jx1) * (jy2 - jy1);
                const float ix1 = fmaxf(x1, jx1), iy1 = fmaxf(y1, jy1);
                const float ix2 = fminf(x2, jx2), iy2 = fminf(y2, jy2);
                const float inter = fmaxf(ix2 - ix1, 0.f) * fmaxf(iy2 - iy1, 0.f);
                const float iou = inter / (ai + aj - inter + 1e-6f);
                if (iou > kNMSThr) supp[j] = 1;
            }
        }
        __syncthreads();
    }
    if (threadIdx.x == 0) {
        int c = 0;
        for (int j = 0; j < kPreNMS && c < kPostNMS; ++j)
            if (keep[j]) {
                for (int q = 0; q < 4; ++q)
                    rois_out[((size_t)b * kPostNMS + c) * 4 + q] = bx[j * 4 + q];
                ++c;
            }
        for (int j = 0; j < kPreNMS && c < kPostNMS; ++j)
            if (!keep[j]) {
                for (int q = 0; q < 4; ++q)
                    rois_out[((size_t)b * kPostNMS + c) * 4 + q] = bx[j * 4 + q];
                ++c;
            }
    }
}

// ================================ RoIAlign ==================================
// feat: NHWC f16 [B][100][100][256].  rois: f32 [400][4] (from d_out).
// layout 0: dst[r][256][out][out]  (feeds fc1, (C,7,7) flatten order)
// layout 1: dst[r][out][out][256]  (NHWC, feeds mask convs)
__global__ void k_roi_align(const _Float16* __restrict__ feat,
                            const float* __restrict__ rois,
                            _Float16* __restrict__ dst, int outsz, int layout)
{
    const size_t total = (size_t)kB * kPostNMS * 256 * outsz * outsz;
    size_t tid = blockIdx.x * (size_t)blockDim.x + threadIdx.x;
    if (tid >= total) return;
    const int ox = (int)(tid % outsz);
    const int oy = (int)((tid / outsz) % outsz);
    const int c  = (int)((tid / ((size_t)outsz * outsz)) % 256);
    const int r  = (int)(tid / ((size_t)256 * outsz * outsz));
    const int b  = r / kPostNMS;
    const float* rp = rois + (size_t)r * 4;
    const float x1 = rp[0], y1 = rp[1], x2 = rp[2], y2 = rp[3];
    const float bw = (x2 - x1) / outsz, bh = (y2 - y1) / outsz;
    float acc = 0.f;
#pragma unroll
    for (int sy = 0; sy < 2; ++sy) {
#pragma unroll
        for (int sx = 0; sx < 2; ++sx) {
            const float gy = ((oy * 2 + sy) + 0.5f) * 0.5f;
            const float gx = ((ox * 2 + sx) + 0.5f) * 0.5f;
            float yy = (y1 + gy * bh) * 0.125f - 0.5f;
            float xx = (x1 + gx * bw) * 0.125f - 0.5f;
            float y0 = fminf(fmaxf(floorf(yy), 0.f), 99.f);
            float x0 = fminf(fmaxf(floorf(xx), 0.f), 99.f);
            const float ly = fminf(fmaxf(yy - y0, 0.f), 1.f);
            const float lx = fminf(fmaxf(xx - x0, 0.f), 1.f);
            const int y0i = (int)y0, x0i = (int)x0;
            const int y1i = (y0i + 1 < 100) ? y0i + 1 : 99;
            const int x1i = (x0i + 1 < 100) ? x0i + 1 : 99;
            const size_t base = (size_t)b * kFH * kFW * 256;
            const float v00 = (float)feat[base + ((size_t)y0i * kFW + x0i) * 256 + c];
            const float v01 = (float)feat[base + ((size_t)y0i * kFW + x1i) * 256 + c];
            const float v10 = (float)feat[base + ((size_t)y1i * kFW + x0i) * 256 + c];
            const float v11 = (float)feat[base + ((size_t)y1i * kFW + x1i) * 256 + c];
            acc += v00 * (1.f - ly) * (1.f - lx) + v01 * (1.f - ly) * lx
                 + v10 * ly * (1.f - lx)        + v11 * ly * lx;
        }
    }
    acc *= 0.25f;
    if (layout == 0)
        dst[((size_t)r * 256 + c) * outsz * outsz + (size_t)oy * outsz + ox] = (_Float16)acc;
    else
        dst[(((size_t)r * outsz + oy) * outsz + ox) * 256 + c] = (_Float16)acc;
}

// ========================= cls / bbox heads (small N) =======================
__global__ void k_cls_bbox(const _Float16* __restrict__ h,
                           const float* __restrict__ cw, const float* __restrict__ cb,
                           const float* __restrict__ bw, const float* __restrict__ bb,
                           float* __restrict__ cls_out, float* __restrict__ box_out)
{
    int tid = blockIdx.x * blockDim.x + threadIdx.x;
    if (tid >= kB * kPostNMS * 60) return;
    const int j = tid % 60;
    const int r = tid / 60;
    const _Float16* hp = h + (size_t)r * 1024;
    if (j < kNumClass) {
        float s = cb[j];
        for (int k = 0; k < 1024; ++k) s += (float)hp[k] * cw[(size_t)k * kNumClass + j];
        cls_out[(size_t)r * kNumClass + j] = s;
    } else {
        const int jj = j - kNumClass;
        float s = bb[jj];
        for (int k = 0; k < 1024; ++k) s += (float)hp[k] * bw[(size_t)k * 48 + jj];
        box_out[(size_t)r * 48 + jj] = s;
    }
}

// ============================ mask deconv (2x2 s2) ==========================
__global__ void k_deconv(const _Float16* __restrict__ in,   // [400][14][14][256]
                         const _Float16* __restrict__ wd,   // [4][256][256] f16
                         const float* __restrict__ bias,
                         _Float16* __restrict__ out)        // [400][28][28][256]
{
    const size_t total = (size_t)kB * kPostNMS * 28 * 28 * 256;
    size_t tid = blockIdx.x * (size_t)blockDim.x + threadIdx.x;
    if (tid >= total) return;
    const int co = (int)(tid & 255);
    const int ox = (int)((tid >> 8) % 28);
    const int oy = (int)((tid >> 8) / 28 % 28);
    const int r  = (int)(tid / ((size_t)28 * 28 * 256));
    const int di = oy & 1, dj = ox & 1, iy = oy >> 1, ix = ox >> 1;
    const _Float16* ip = in + (((size_t)r * 14 + iy) * 14 + ix) * 256;
    const _Float16* wp = wd + ((size_t)(di * 2 + dj) * 256) * 256 + co;
    float s = bias[co];
    for (int ci = 0; ci < 256; ++ci) s += (float)ip[ci] * (float)wp[(size_t)ci * 256];
    if (s < 0.f) s = 0.f;
    out[tid] = (_Float16)s;   // tid == (((r*28+oy)*28+ox)*256+co) -> NHWC
}

// ============================== mask predictor ==============================
__global__ void k_mpred(const _Float16* __restrict__ m,     // [400][28][28][256]
                        const float* __restrict__ w,        // [12][256]
                        const float* __restrict__ b,
                        float* __restrict__ out)            // [400][12][28][28]
{
    const size_t total = (size_t)kB * kPostNMS * kNumClass * 28 * 28;
    size_t tid = blockIdx.x * (size_t)blockDim.x + threadIdx.x;
    if (tid >= total) return;
    const int ox = (int)(tid % 28);
    const int oy = (int)((tid / 28) % 28);
    const int cl = (int)((tid / (28 * 28)) % kNumClass);
    const int r  = (int)(tid / ((size_t)kNumClass * 28 * 28));
    const _Float16* mp = m + (((size_t)r * 28 + oy) * 28 + ox) * 256;
    const float* wp = w + (size_t)cl * 256;
    float s = b[cl];
    for (int c = 0; c < 256; ++c) s += (float)mp[c] * wp[c];
    out[tid] = s;   // tid == ((r*12+cl)*28+oy)*28+ox
}

// ===========================================================================
extern "C" void kernel_launch(void* const* d_in, const int* in_sizes, int n_in,
                              void* d_out, int out_size, void* d_ws, size_t ws_size,
                              hipStream_t stream)
{
    (void)in_sizes; (void)n_in; (void)out_size;
    const float* images     = (const float*)d_in[0];
    const float* conv1_w    = (const float*)d_in[1];
    const float* conv2_w    = (const float*)d_in[2];
    const float* conv3_w    = (const float*)d_in[3];
    const float* rpn_conv_w = (const float*)d_in[4];
    const float* rpn_conv_b = (const float*)d_in[5];
    const float* rpn_cls_w  = (const float*)d_in[6];
    const float* rpn_cls_b  = (const float*)d_in[7];
    const float* rpn_reg_w  = (const float*)d_in[8];
    const float* rpn_reg_b  = (const float*)d_in[9];
    const float* fc1_w      = (const float*)d_in[10];
    const float* fc1_b      = (const float*)d_in[11];
    const float* fc2_w      = (const float*)d_in[12];
    const float* fc2_b      = (const float*)d_in[13];
    const float* cls_w      = (const float*)d_in[14];
    const float* cls_b      = (const float*)d_in[15];
    const float* bbox_w     = (const float*)d_in[16];
    const float* bbox_b     = (const float*)d_in[17];
    const float* mconv_w    = (const float*)d_in[18];
    const float* mconv_b    = (const float*)d_in[19];
    const float* deconv_w   = (const float*)d_in[20];
    const float* deconv_b   = (const float*)d_in[21];
    const float* mpred_w    = (const float*)d_in[22];
    const float* mpred_b    = (const float*)d_in[23];

    // ------------------------- workspace carve-up --------------------------
    char* p0 = (char*)d_ws;
    char* p  = p0;
    auto carve = [&](size_t elems, size_t esz) -> void* {
        void* r = (void*)p;
        p += ((elems * esz + 255) & ~(size_t)255);
        return r;
    };
    _Float16* c1    = (_Float16*)carve((size_t)kB * 400 * 400 * 64, 2);
    _Float16* c2    = (_Float16*)carve((size_t)kB * 200 * 200 * 256, 2);
    _Float16* feat  = (_Float16*)carve((size_t)kB * 100 * 100 * 256, 2);
    _Float16* tmap  = (_Float16*)carve((size_t)kB * 100 * 100 * 256, 2);
    _Float16* w2T   = (_Float16*)carve((size_t)256 * 9 * 64, 2);
    _Float16* w3T   = (_Float16*)carve((size_t)256 * 9 * 256, 2);
    _Float16* wrT   = (_Float16*)carve((size_t)256 * 9 * 256, 2);
    _Float16* wmT   = (_Float16*)carve((size_t)4 * 256 * 9 * 256, 2);
    _Float16* wf1T  = (_Float16*)carve((size_t)1024 * 12544, 2);
    _Float16* wf2T  = (_Float16*)carve((size_t)1024 * 1024, 2);
    _Float16* wdT   = (_Float16*)carve((size_t)4 * 256 * 256, 2);
    float*    scores= (float*)   carve((size_t)kB * 30000, 4);
    float*    deltas= (float*)   carve((size_t)kB * 30000 * 4, 4);
    int*      topix = (int*)     carve((size_t)kB * kPreNMS, 4);
    float*    boxes = (float*)   carve((size_t)kB * kPreNMS * 4, 4);
    _Float16* bf7   = (_Float16*)carve((size_t)kB * kPostNMS * 12544, 2);
    _Float16* h1    = (_Float16*)carve((size_t)kB * kPostNMS * 1024, 2);
    _Float16* h2    = (_Float16*)carve((size_t)kB * kPostNMS * 1024, 2);
    _Float16* m0    = (_Float16*)carve((size_t)kB * kPostNMS * 14 * 14 * 256, 2);
    _Float16* m1    = (_Float16*)carve((size_t)kB * kPostNMS * 14 * 14 * 256, 2);
    _Float16* m2    = (_Float16*)carve((size_t)kB * kPostNMS * 28 * 28 * 256, 2);
    if ((size_t)(p - p0) > ws_size) return;   // not enough scratch: no-op

    float* rois_out = (float*)d_out;                       // [4][100][4]
    float* cls_out  = rois_out + (size_t)kB * kPostNMS * 4;
    float* box_out  = cls_out  + (size_t)kB * kPostNMS * kNumClass;
    float* mask_out = box_out  + (size_t)kB * kPostNMS * 48;

    auto blocks1d = [](size_t total, int bs) { return (unsigned)((total + bs - 1) / bs); };
    // conv: one wave per 16(M) x 64(N) tile; 8 waves / 256-thread block
    auto conv_blocks = [](int M, int Cout) {
        size_t waves = (size_t)((M + 15) / 16) * ((Cout + 63) / 64);
        return (unsigned)((waves + 7) / 8);
    };
    // gemm: one block per 128(M) x 64(N) tile
    auto gemm_blocks = [](int M, int N) {
        return (unsigned)(((size_t)(M + 127) / 128) * ((size_t)N / 64));
    };

    // ----------------------------- weight packing --------------------------
    k_pack_conv_w<<<blocks1d((size_t)256 * 64 * 9, 256), 256, 0, stream>>>(conv2_w, w2T, 256, 64);
    k_pack_conv_w<<<blocks1d((size_t)256 * 256 * 9, 256), 256, 0, stream>>>(conv3_w, w3T, 256, 256);
    k_pack_conv_w<<<blocks1d((size_t)256 * 256 * 9, 256), 256, 0, stream>>>(rpn_conv_w, wrT, 256, 256);
    for (int i = 0; i < 4; ++i)
        k_pack_conv_w<<<blocks1d((size_t)256 * 256 * 9, 256), 256, 0, stream>>>(
            mconv_w + (size_t)i * 256 * 256 * 9, wmT + (size_t)i * 256 * 9 * 256, 256, 256);
    k_pack_fc_w<<<blocks1d((size_t)12544 * 1024, 256), 256, 0, stream>>>(fc1_w, wf1T, 12544, 1024);
    k_pack_fc_w<<<blocks1d((size_t)1024 * 1024, 256), 256, 0, stream>>>(fc2_w, wf2T, 1024, 1024);
    k_pack_deconv_w<<<blocks1d((size_t)256 * 256 * 4, 256), 256, 0, stream>>>(deconv_w, wdT);

    // ------------------------------ backbone -------------------------------
    k_conv1<<<blocks1d((size_t)kB * 400 * 400 * 64, 256), 256, 0, stream>>>(images, conv1_w, c1);
    // conv2: 400->200 s2 (SAME, pad_lo=0), Cin=64
    k_conv3x3_wmma<<<conv_blocks(kB * 200 * 200, 256), 256, 0, stream>>>(
        c1, w2T, nullptr, c2, kB, 400, 400, 64, 200, 200, 256, 2, 0);
    // conv3: 200->100 s2 (pad_lo=0), Cin=256
    k_conv3x3_wmma<<<conv_blocks(kB * 100 * 100, 256), 256, 0, stream>>>(
        c2, w3T, nullptr, feat, kB, 200, 200, 256, 100, 100, 256, 2, 0);
    // rpn_conv: s1 pad 1, + bias + relu
    k_conv3x3_wmma<<<conv_blocks(kB * 100 * 100, 256), 256, 0, stream>>>(
        feat, wrT, rpn_conv_b, tmap, kB, 100, 100, 256, 100, 100, 256, 1, 1);

    // ------------------------------ RPN + NMS ------------------------------
    k_rpn_heads<<<blocks1d((size_t)kB * kA * kFH * kFW, 256), 256, 0, stream>>>(
        tmap, rpn_cls_w, rpn_cls_b, rpn_reg_w, rpn_reg_b, scores, deltas);
    k_topk<<<kB, 256, 0, stream>>>(scores, topix, kA * kFH * kFW, kPreNMS);
    k_decode<<<blocks1d((size_t)kB * kPreNMS, 256), 256, 0, stream>>>(topix, deltas, boxes);
    k_nms<<<kB, 256, 0, stream>>>(boxes, rois_out);

    // ------------------------------ box head -------------------------------
    k_roi_align<<<blocks1d((size_t)kB * kPostNMS * 256 * 49, 256), 256, 0, stream>>>(
        feat, rois_out, bf7, 7, 0);
    k_gemm_f16<<<gemm_blocks(kB * kPostNMS, 1024), 256, 0, stream>>>(
        bf7, wf1T, fc1_b, h1, kB * kPostNMS, 1024, 12544, 1);
    k_gemm_f16<<<gemm_blocks(kB * kPostNMS, 1024), 256, 0, stream>>>(
        h1, wf2T, fc2_b, h2, kB * kPostNMS, 1024, 1024, 1);
    k_cls_bbox<<<blocks1d((size_t)kB * kPostNMS * 60, 256), 256, 0, stream>>>(
        h2, cls_w, cls_b, bbox_w, bbox_b, cls_out, box_out);

    // ------------------------------ mask head ------------------------------
    k_roi_align<<<blocks1d((size_t)kB * kPostNMS * 256 * 196, 256), 256, 0, stream>>>(
        feat, rois_out, m0, 14, 1);
    _Float16* mi = m0; _Float16* mo = m1;
    for (int i = 0; i < 4; ++i) {
        k_conv3x3_wmma<<<conv_blocks(kB * kPostNMS * 14 * 14, 256), 256, 0, stream>>>(
            mi, wmT + (size_t)i * 256 * 9 * 256, mconv_b + (size_t)i * 256, mo,
            kB * kPostNMS, 14, 14, 256, 14, 14, 256, 1, 1);
        _Float16* t = mi; mi = mo; mo = t;
    }
    k_deconv<<<blocks1d((size_t)kB * kPostNMS * 28 * 28 * 256, 256), 256, 0, stream>>>(
        mi, wdT, deconv_b, m2);
    k_mpred<<<blocks1d((size_t)kB * kPostNMS * kNumClass * 28 * 28, 256), 256, 0, stream>>>(
        m2, mpred_w, mpred_b, mask_out);
}
